// SwinTransformerBlock_52939766891033
// MI455X (gfx1250) — compile-verified, matrix operand order checked by reading, not run on hardware
//
#include <hip/hip_runtime.h>
#include <hip/hip_bf16.h>
#include <math.h>

typedef __attribute__((ext_vector_type(16))) _Float16 v16h;
typedef __attribute__((ext_vector_type(8)))  _Float16 v8h;
typedef __attribute__((ext_vector_type(8)))  float    v8f;

// ---------------- problem constants ----------------
#define BB    32
#define HH    56
#define WW    56
#define CC    192
#define WS7   7
#define NHEAD 6
#define HD    32
#define NTOK  49            // tokens per window
#define NWIN  64            // windows per image (8x8)
#define BWIN  (BB*NWIN)     // 2048 global windows
#define MTOT  (BWIN*NTOK)   // 100352 tokens
#define MLPC  768

// GEMM tiling
#define BM 128
#define BN 64
#define BK 32

// ---------------------------------------------------------------------------
// Fragment loader (wave32, 16-bit A/B layout per CDNA5 ISA 7.12.2):
// lane = l : row/col = l&15 ; k chunks: halves 0..7 -> k = (l<16?0:8)+h,
// halves 8..15 -> k = (l<16?16:24)+(h-8). Two 16-byte LDS loads per fragment.
// ---------------------------------------------------------------------------
__device__ __forceinline__ v16h frag_ld(const _Float16* base, int ldh) {
  int lane = threadIdx.x & 31;
  const _Float16* p = base + (size_t)(lane & 15) * ldh + ((lane >> 4) & 1) * 8;
  v8h lo = *(const v8h*)(p);
  v8h hi = *(const v8h*)(p + 16);
  return __builtin_shufflevector(lo, hi, 0,1,2,3,4,5,6,7,8,9,10,11,12,13,14,15);
}

__device__ __forceinline__ v8f wmma_f16(v16h a, v16h b, v8f c) {
  return __builtin_amdgcn_wmma_f32_16x16x32_f16(false, a, false, b, (short)0, c,
                                                false, false);
}

// ---------------------------------------------------------------------------
// CDNA5 async copy: GLOBAL_LOAD_ASYNC_TO_LDS_B128 (ASYNCcnt-tracked,
// cache -> LDS with no VGPR staging). Per-lane: 16B from gptr -> LDS at
// lds byte-offset (low 32 bits of the generic shared pointer).
// ---------------------------------------------------------------------------
__device__ __forceinline__ void async_copy_b128(unsigned lds_off,
                                                const _Float16* gptr) {
  asm volatile("global_load_async_to_lds_b128 %0, %1, off"
               :
               : "v"(lds_off), "v"((unsigned long long)(size_t)gptr)
               : "memory");
}

template <int N>
__device__ __forceinline__ void wait_async() {
  asm volatile("s_wait_asynccnt %0" : : "n"(N) : "memory");
}

__device__ __forceinline__ unsigned lds_off32(const void* p) {
  return (unsigned)(size_t)p;   // flat->LDS: addr[31:0] is the LDS offset
}

// ---------------------------------------------------------------------------
// small helper kernels
// ---------------------------------------------------------------------------
__global__ void cvt_f16_kernel(const float* __restrict__ src,
                               _Float16* __restrict__ dst, int n) {
  int i = blockIdx.x * 256 + threadIdx.x;
  if (i < n) dst[i] = (_Float16)src[i];
}

__global__ void bias_table_kernel(const float* __restrict__ rpb,
                                  float* __restrict__ biasT) {
  int idx = blockIdx.x * 256 + threadIdx.x;     // NHEAD*49*49 = 14406
  if (idx >= NHEAD * NTOK * NTOK) return;
  int h = idx / (NTOK * NTOK);
  int rem = idx % (NTOK * NTOK);
  int n = rem / NTOK, m = rem % NTOK;
  int i1 = n / WS7, j1 = n % WS7, i2 = m / WS7, j2 = m % WS7;
  int rel = (i1 - i2 + WS7 - 1) * (2 * WS7 - 1) + (j1 - j2 + WS7 - 1);
  biasT[idx] = rpb[rel * NHEAD + h];
}

// ---------------------------------------------------------------------------
// LayerNorm, one wave (32 lanes) per token, 6 channels per lane
// ---------------------------------------------------------------------------
__device__ __forceinline__ void ln_row(const float* __restrict__ src,
                                       const float* __restrict__ w,
                                       const float* __restrict__ b,
                                       _Float16* __restrict__ dst, int lane) {
  float v[6];
  float s = 0.f;
#pragma unroll
  for (int i = 0; i < 6; ++i) { v[i] = src[lane + 32 * i]; s += v[i]; }
#pragma unroll
  for (int o = 16; o > 0; o >>= 1) s += __shfl_xor(s, o, 32);
  float mean = s * (1.0f / 192.0f);
  float q = 0.f;
#pragma unroll
  for (int i = 0; i < 6; ++i) { float d = v[i] - mean; q += d * d; }
#pragma unroll
  for (int o = 16; o > 0; o >>= 1) q += __shfl_xor(q, o, 32);
  float inv = rsqrtf(q * (1.0f / 192.0f) + 1e-5f);
#pragma unroll
  for (int i = 0; i < 6; ++i) {
    int c = lane + 32 * i;
    dst[c] = (_Float16)((v[i] - mean) * inv * w[c] + b[c]);
  }
}

// LN1 + cyclic shift + window partition: output in window order, f16
__global__ __launch_bounds__(256) void ln1_shift_kernel(
    const float* __restrict__ x, const float* __restrict__ w,
    const float* __restrict__ b, _Float16* __restrict__ xw) {
  int token = blockIdx.x * 8 + (threadIdx.x >> 5);
  int lane = threadIdx.x & 31;
  int win = token / NTOK, n = token % NTOK;
  int bi = win >> 6, wloc = win & 63;
  int hs = (wloc >> 3) * WS7 + n / WS7;
  int ws = (wloc & 7) * WS7 + n % WS7;
  int sh = hs + 3; if (sh >= HH) sh -= HH;   // xs[h] = x[(h+3)%56]
  int sw = ws + 3; if (sw >= WW) sw -= WW;
  const float* src = x + ((size_t)bi * (HH * WW) + sh * WW + sw) * CC;
  ln_row(src, w, b, xw + (size_t)token * CC, lane);
}

// LN2: spatial order in/out
__global__ __launch_bounds__(256) void ln2_kernel(
    const float* __restrict__ x1, const float* __restrict__ w,
    const float* __restrict__ b, _Float16* __restrict__ h16) {
  int token = blockIdx.x * 8 + (threadIdx.x >> 5);
  int lane = threadIdx.x & 31;
  ln_row(x1 + (size_t)token * CC, w, b, h16 + (size_t)token * CC, lane);
}

// ---------------------------------------------------------------------------
// Generic WMMA GEMM mainloop:  D[128x64] += A[128xK] * Bt[64xK]^T
// 256 threads = 8 waves, wave (wm 0..3, wn 0..1), each wave 2x2 16x16 tiles.
// Double-buffered LDS, tiles fetched by GLOBAL_LOAD_ASYNC_TO_LDS_B128:
// tile k+1 streams into the alternate buffer while WMMAs consume tile k.
// Each thread issues 3 async ops per k-step (2 A-chunks + 1 B-chunk).
// ---------------------------------------------------------------------------
__device__ __forceinline__ void issue_tile(const _Float16* __restrict__ A,
                                           const _Float16* __restrict__ Bt,
                                           int K, int m0, int n0, int k0,
                                           _Float16* sA, _Float16* sB) {
  const int tid = threadIdx.x;
#pragma unroll
  for (int i = 0; i < 2; ++i) {
    int c = tid * 2 + i;                 // 0..511 : A tile 128x32
    int r = c >> 2, kk = (c & 3) * 8;
    async_copy_b128(lds_off32(sA + r * BK + kk),
                    A + (size_t)(m0 + r) * K + k0 + kk);
  }
  {
    int r = tid >> 2, kk = (tid & 3) * 8;  // B tile 64x32
    async_copy_b128(lds_off32(sB + r * BK + kk),
                    Bt + (size_t)(n0 + r) * K + k0 + kk);
  }
}

__device__ __forceinline__ void gemm_mainloop(
    const _Float16* __restrict__ A, const _Float16* __restrict__ Bt, int K,
    int m0, int n0, _Float16* sA, _Float16* sB, v8f acc[2][2]) {
  const int wid = threadIdx.x >> 5, wm = wid & 3, wn = wid >> 2;
  const int nk = K / BK;

  issue_tile(A, Bt, K, m0, n0, 0, sA, sB);

  for (int k = 0; k < nk; ++k) {
    int cur = k & 1;
    if (k + 1 < nk) {
      // alternate buffer was last read in iter k-1; its trailing barrier
      // already passed, so streaming into it now is safe.
      issue_tile(A, Bt, K, m0, n0, (k + 1) * BK, sA + (cur ^ 1) * (BM * BK),
                 sB + (cur ^ 1) * (BN * BK));
      wait_async<3>();   // our 3 ops for tile k complete (in-order)
    } else {
      wait_async<0>();
    }
    __syncthreads();     // all waves' tile-k data visible in LDS

    _Float16* cA = sA + cur * (BM * BK);
    _Float16* cB = sB + cur * (BN * BK);
    v16h af[2], bf[2];
#pragma unroll
    for (int tm = 0; tm < 2; ++tm)
      af[tm] = frag_ld(cA + (wm * 32 + tm * 16) * BK, BK);
#pragma unroll
    for (int tn = 0; tn < 2; ++tn)
      bf[tn] = frag_ld(cB + (wn * 32 + tn * 16) * BK, BK);
#pragma unroll
    for (int tm = 0; tm < 2; ++tm)
#pragma unroll
      for (int tn = 0; tn < 2; ++tn)
        acc[tm][tn] = wmma_f16(af[tm], bf[tn], acc[tm][tn]);
    __syncthreads();     // buffer free for issue at iter k+2
  }
}

#define GEMM_PROLOG(KDIM)                                     \
  __shared__ _Float16 sA[2 * BM * BK];                        \
  __shared__ _Float16 sB[2 * BN * BK];                        \
  v8f acc[2][2];                                              \
  _Pragma("unroll") for (int tm = 0; tm < 2; ++tm)            \
      _Pragma("unroll") for (int tn = 0; tn < 2; ++tn)        \
          _Pragma("unroll") for (int r = 0; r < 8; ++r)       \
              acc[tm][tn][r] = 0.f;                           \
  int m0 = blockIdx.y * BM, n0 = blockIdx.x * BN;             \
  gemm_mainloop(A, Bt, (KDIM), m0, n0, sA, sB, acc);          \
  int lane = threadIdx.x & 31;                                \
  int wid = threadIdx.x >> 5, wm = wid & 3, wn = wid >> 2;

#define GEMM_FOREACH                                          \
  _Pragma("unroll") for (int tm = 0; tm < 2; ++tm)            \
  _Pragma("unroll") for (int tn = 0; tn < 2; ++tn) {          \
    int row0 = m0 + wm * 32 + tm * 16 + ((lane >> 4) << 3);   \
    int col = n0 + wn * 32 + tn * 16 + (lane & 15);           \
    _Pragma("unroll") for (int r = 0; r < 8; ++r) {           \
      int row = row0 + r;                                     \
      float val = acc[tm][tn][r];

#define GEMM_END } }

// ---- QKV GEMM: N=576, epilogue scatters into per-(window,head) q/k/v ------
__global__ __launch_bounds__(256) void gemm_qkv_kernel(
    const _Float16* __restrict__ A, const _Float16* __restrict__ Bt,
    const float* __restrict__ bias, _Float16* __restrict__ qo,
    _Float16* __restrict__ ko, _Float16* __restrict__ vo) {
  GEMM_PROLOG(CC)
  GEMM_FOREACH
    float v = val + bias[col];
    int seg = col / CC, ch = col % CC;
    int hh = ch >> 5, d = ch & 31;
    if (seg == 0) v *= 0.17677669529663687f;   // HD^-0.5 folded into q
    int win = row / NTOK, n = row % NTOK;
    size_t idx = (((size_t)win * NHEAD + hh) * NTOK + n) * HD + d;
    _Float16* dst = (seg == 0) ? qo : ((seg == 1) ? ko : vo);
    dst[idx] = (_Float16)v;
  GEMM_END
}

// ---- proj GEMM + window reverse + un-shift + residual ---------------------
__global__ __launch_bounds__(256) void gemm_proj_kernel(
    const _Float16* __restrict__ A, const _Float16* __restrict__ Bt,
    const float* __restrict__ bias, const float* __restrict__ xin,
    float* __restrict__ x1) {
  GEMM_PROLOG(CC)
  GEMM_FOREACH
    float v = val + bias[col];
    int win = row / NTOK, n = row % NTOK;
    int bi = win >> 6, wloc = win & 63;
    int hs = (wloc >> 3) * WS7 + n / WS7;
    int ws = (wloc & 7) * WS7 + n % WS7;
    int oh = hs + 3; if (oh >= HH) oh -= HH;   // final[(hs+3)%56]
    int ow = ws + 3; if (ow >= WW) ow -= WW;
    size_t idx = ((size_t)bi * (HH * WW) + oh * WW + ow) * CC + col;
    x1[idx] = xin[idx] + v;
  GEMM_END
}

// ---- fc1 GEMM + exact GELU ------------------------------------------------
__global__ __launch_bounds__(256) void gemm_fc1_kernel(
    const _Float16* __restrict__ A, const _Float16* __restrict__ Bt,
    const float* __restrict__ bias, _Float16* __restrict__ h2) {
  GEMM_PROLOG(CC)
  GEMM_FOREACH
    float v = val + bias[col];
    float g = 0.5f * v * (1.0f + erff(v * 0.70710678118654752f));
    h2[(size_t)row * MLPC + col] = (_Float16)g;
  GEMM_END
}

// ---- fc2 GEMM + bias + residual -> output ---------------------------------
__global__ __launch_bounds__(256) void gemm_fc2_kernel(
    const _Float16* __restrict__ A, const _Float16* __restrict__ Bt,
    const float* __restrict__ bias, const float* __restrict__ x1,
    float* __restrict__ out) {
  GEMM_PROLOG(MLPC)
  GEMM_FOREACH
    size_t idx = (size_t)row * CC + col;
    out[idx] = x1[idx] + val + bias[col];
  GEMM_END
}

// ---------------------------------------------------------------------------
// Attention: one block per (window, head). 128 threads = 4 waves.
// 49 padded to 64. S = q*k^T (K=32, single WMMA per tile), +bias+mask,
// softmax, P(f16) @ V (V transposed in LDS) -> out rows.
// ---------------------------------------------------------------------------
__global__ __launch_bounds__(128) void attn_kernel(
    const _Float16* __restrict__ qb, const _Float16* __restrict__ kb,
    const _Float16* __restrict__ vb, const float* __restrict__ biasT,
    const float* __restrict__ mask, _Float16* __restrict__ outb) {
  __shared__ _Float16 sQ[64 * 32];
  __shared__ _Float16 sK[64 * 32];
  __shared__ _Float16 sVT[32 * 64];
  __shared__ float    sS[64 * 64];
  __shared__ _Float16 sP[64 * 64];

  int bh = blockIdx.x;                 // win*NHEAD + h
  int win = bh / NHEAD, h = bh % NHEAD;
  int wimg = win & 63;                 // window id within image -> mask row
  int tid = threadIdx.x, lane = tid & 31, wv = tid >> 5;

  const _Float16* qg = qb + (size_t)bh * NTOK * HD;
  const _Float16* kg = kb + (size_t)bh * NTOK * HD;
  const _Float16* vg = vb + (size_t)bh * NTOK * HD;

  v8h zero8;
#pragma unroll
  for (int j = 0; j < 8; ++j) zero8[j] = (_Float16)0.f;

  // load q,k (rows 49..63 zero)
  for (int i = tid; i < 256; i += 128) {
    int r = i >> 2, kk = (i & 3) * 8;
    v8h vq = zero8, vk = zero8;
    if (r < NTOK) {
      vq = *(const v8h*)(qg + r * HD + kk);
      vk = *(const v8h*)(kg + r * HD + kk);
    }
    *(v8h*)(sQ + r * 32 + kk) = vq;
    *(v8h*)(sK + r * 32 + kk) = vk;
  }
  // load v transposed: sVT[d][n]
  for (int i = tid; i < 32 * 64; i += 128) {
    int d = i >> 6, n = i & 63;
    sVT[d * 64 + n] = (n < NTOK) ? vg[n * HD + d] : (_Float16)0.f;
  }
  __syncthreads();

  // S = q * k^T  : wave wv owns m-tile wv, 4 n-tiles
  v8f accS[4];
#pragma unroll
  for (int t = 0; t < 4; ++t)
#pragma unroll
    for (int r = 0; r < 8; ++r) accS[t][r] = 0.f;
  {
    v16h aq = frag_ld(sQ + (wv * 16) * 32, 32);
#pragma unroll
    for (int nt = 0; nt < 4; ++nt) {
      v16h bk = frag_ld(sK + (nt * 16) * 32, 32);
      accS[nt] = wmma_f16(aq, bk, accS[nt]);
    }
  }
  // dump + bias + mask (+ -inf padding)
  {
    const float* biasH = biasT + (size_t)h * NTOK * NTOK;
    const float* maskW = mask + (size_t)wimg * NTOK * NTOK;
    int colb = lane & 15;
    int rowb = wv * 16 + ((lane >> 4) << 3);
#pragma unroll
    for (int nt = 0; nt < 4; ++nt)
#pragma unroll
      for (int r = 0; r < 8; ++r) {
        int row = rowb + r, col = nt * 16 + colb;
        float s = -1e30f;
        if (row < NTOK && col < NTOK)
          s = accS[nt][r] + biasH[row * NTOK + col] + maskW[row * NTOK + col];
        sS[row * 64 + col] = s;
      }
  }
  __syncthreads();

  // softmax per row (rows 0..48), P in f16, padded entries zero
  if (tid < 64) {
    int row = tid;
    if (row < NTOK) {
      float m = -1e30f;
      for (int c = 0; c < NTOK; ++c) m = fmaxf(m, sS[row * 64 + c]);
      float den = 0.f;
      for (int c = 0; c < NTOK; ++c) den += __expf(sS[row * 64 + c] - m);
      float inv = 1.0f / den;
      for (int c = 0; c < 64; ++c)
        sP[row * 64 + c] =
            (c < NTOK) ? (_Float16)(__expf(sS[row * 64 + c] - m) * inv)
                       : (_Float16)0.f;
    } else {
      for (int c = 0; c < 64; ++c) sP[row * 64 + c] = (_Float16)0.f;
    }
  }
  __syncthreads();

  // out = P @ V : wave wv m-tile, 2 n-tiles (HD=32), K=64 -> 2 k-steps
  v8f accO[2];
#pragma unroll
  for (int t = 0; t < 2; ++t)
#pragma unroll
    for (int r = 0; r < 8; ++r) accO[t][r] = 0.f;
#pragma unroll
  for (int ks = 0; ks < 2; ++ks) {
    v16h ap = frag_ld(sP + (wv * 16) * 64 + ks * 32, 64);
#pragma unroll
    for (int nt = 0; nt < 2; ++nt) {
      v16h bv = frag_ld(sVT + (nt * 16) * 64 + ks * 32, 64);
      accO[nt] = wmma_f16(ap, bv, accO[nt]);
    }
  }
  // write out rows [win*49+n][h*32+d] as f16 (feeds proj GEMM)
  {
    int colb = lane & 15;
    int rowb = wv * 16 + ((lane >> 4) << 3);
#pragma unroll
    for (int nt = 0; nt < 2; ++nt)
#pragma unroll
      for (int r = 0; r < 8; ++r) {
        int n = rowb + r, d = nt * 16 + colb;
        if (n < NTOK)
          outb[((size_t)win * NTOK + n) * CC + h * HD + d] =
              (_Float16)accO[nt][r];
      }
  }
}

// ---------------------------------------------------------------------------
extern "C" void kernel_launch(void* const* d_in, const int* in_sizes, int n_in,
                              void* d_out, int out_size, void* d_ws,
                              size_t ws_size, hipStream_t stream) {
  const float* x      = (const float*)d_in[0];
  const float* amask  = (const float*)d_in[1];
  const float* n1w    = (const float*)d_in[2];
  const float* n1b    = (const float*)d_in[3];
  const float* qkv_w  = (const float*)d_in[4];
  const float* qkv_b  = (const float*)d_in[5];
  const float* rpb    = (const float*)d_in[6];
  const float* proj_w = (const float*)d_in[7];
  const float* proj_b = (const float*)d_in[8];
  const float* n2w    = (const float*)d_in[9];
  const float* n2b    = (const float*)d_in[10];
  const float* fc1_w  = (const float*)d_in[11];
  const float* fc1_b  = (const float*)d_in[12];
  const float* fc2_w  = (const float*)d_in[13];
  const float* fc2_b  = (const float*)d_in[14];
  float* out = (float*)d_out;

  char* ws = (char*)d_ws;
  size_t off = 0;
  auto alloc = [&](size_t bytes) -> void* {
    void* p = ws + off;
    off = (off + bytes + 255) & ~(size_t)255;
    return p;
  };

  _Float16* wq16  = (_Float16*)alloc((size_t)3 * CC * CC * 2);
  _Float16* wp16  = (_Float16*)alloc((size_t)CC * CC * 2);
  _Float16* wf116 = (_Float16*)alloc((size_t)MLPC * CC * 2);
  _Float16* wf216 = (_Float16*)alloc((size_t)CC * MLPC * 2);
  float*    biasT = (float*)alloc((size_t)NHEAD * NTOK * NTOK * 4);
  _Float16* xw16  = (_Float16*)alloc((size_t)MTOT * CC * 2);
  _Float16* q16   = (_Float16*)alloc((size_t)BWIN * NHEAD * NTOK * HD * 2);
  _Float16* k16   = (_Float16*)alloc((size_t)BWIN * NHEAD * NTOK * HD * 2);
  _Float16* v16   = (_Float16*)alloc((size_t)BWIN * NHEAD * NTOK * HD * 2);
  _Float16* ao16  = (_Float16*)alloc((size_t)MTOT * CC * 2);
  float*    x1    = (float*)alloc((size_t)MTOT * CC * 4);
  _Float16* h16   = (_Float16*)alloc((size_t)MTOT * CC * 2);
  _Float16* h216  = (_Float16*)alloc((size_t)MTOT * MLPC * 2);

  // weight conversions
  cvt_f16_kernel<<<(3 * CC * CC + 255) / 256, 256, 0, stream>>>(qkv_w, wq16,
                                                                3 * CC * CC);
  cvt_f16_kernel<<<(CC * CC + 255) / 256, 256, 0, stream>>>(proj_w, wp16,
                                                            CC * CC);
  cvt_f16_kernel<<<(MLPC * CC + 255) / 256, 256, 0, stream>>>(fc1_w, wf116,
                                                              MLPC * CC);
  cvt_f16_kernel<<<(CC * MLPC + 255) / 256, 256, 0, stream>>>(fc2_w, wf216,
                                                              CC * MLPC);
  bias_table_kernel<<<(NHEAD * NTOK * NTOK + 255) / 256, 256, 0, stream>>>(
      rpb, biasT);

  // LN1 + shift + window partition
  ln1_shift_kernel<<<MTOT / 8, 256, 0, stream>>>(x, n1w, n1b, xw16);

  // QKV
  gemm_qkv_kernel<<<dim3(3 * CC / BN, MTOT / BM), 256, 0, stream>>>(
      xw16, wq16, qkv_b, q16, k16, v16);

  // attention
  attn_kernel<<<BWIN * NHEAD, 128, 0, stream>>>(q16, k16, v16, biasT, amask,
                                                ao16);

  // proj + reverse-shift + residual
  gemm_proj_kernel<<<dim3(CC / BN, MTOT / BM), 256, 0, stream>>>(
      ao16, wp16, proj_b, x, x1);

  // LN2
  ln2_kernel<<<MTOT / 8, 256, 0, stream>>>(x1, n2w, n2b, h16);

  // fc1 + GELU
  gemm_fc1_kernel<<<dim3(MLPC / BN, MTOT / BM), 256, 0, stream>>>(
      h16, wf116, fc1_b, h216);

  // fc2 + residual -> out
  gemm_fc2_kernel<<<dim3(CC / BN, MTOT / BM), 256, 0, stream>>>(
      h216, wf216, fc2_b, x1, out);
}
